// CausalSelfAttention_7713761264027
// MI455X (gfx1250) — matrix operand order repaired and checked by reading, not compiled, not run on hardware
//
#include <hip/hip_runtime.h>
#include <stdint.h>

typedef __attribute__((ext_vector_type(16))) __bf16          v16bf;
typedef __attribute__((ext_vector_type(8)))  float           v8f;
typedef __attribute__((ext_vector_type(4)))  float           v4f;
typedef __attribute__((ext_vector_type(16))) unsigned short  v16u;
typedef __attribute__((ext_vector_type(8)))  unsigned short  v8u;
typedef __attribute__((ext_vector_type(4)))  unsigned short  v4u;

__device__ __forceinline__ unsigned short f32_to_bf16(float f) {
  uint32_t u = __builtin_bit_cast(uint32_t, f);
  return (unsigned short)((u + 0x7FFFu + ((u >> 16) & 1u)) >> 16);  // RNE
}
__device__ __forceinline__ v16bf as_bf16(v16u v) { return __builtin_bit_cast(v16bf, v); }

// ---------------------------------------------------------------------------
// One-time packers: fp32 -> bf16 (keeps conversions out of the GEMM hot loop)
// ---------------------------------------------------------------------------
__global__ __launch_bounds__(256) void cvt_f32_to_bf16_vec(
    const float* __restrict__ in, unsigned short* __restrict__ out, int n4)
{
  const int i = blockIdx.x * 256 + threadIdx.x;
  if (i < n4) {
    const v4f v = ((const v4f*)in)[i];
    v4u o;
#pragma unroll
    for (int e = 0; e < 4; ++e) o[e] = f32_to_bf16(v[e]);
    ((v4u*)out)[i] = o;
  }
}

// W [K,N] fp32  ->  WT [N,K] bf16  (so GEMM B-operands are contiguous loads)
__global__ __launch_bounds__(256) void transpose_cvt_bf16(
    const float* __restrict__ W, unsigned short* __restrict__ WT, int K, int N)
{
  const int idx = blockIdx.x * 256 + threadIdx.x;
  if (idx < K * N) {
    const int k = idx / N, n = idx % N;
    WT[(size_t)n * K + k] = f32_to_bf16(W[idx]);
  }
}

// ---------------------------------------------------------------------------
// GEMM: C[M,N] = A[M,K] * B[K,N], A bf16 [M,K], B given as BT bf16 [N,K].
// One wave computes a (16*MT) x (16*NT) tile: MT*NT WMMAs per 32-deep k-step.
// OUT_MODE: 0 = bf16 row-major [M,N], 1 = bf16 transposed [N,M], 2 = f32 [M,N]
// ---------------------------------------------------------------------------
template <int OUT_MODE, int MT, int NT>
__global__ __launch_bounds__(32) void gemm_bf16_wmma(
    const unsigned short* __restrict__ A,
    const unsigned short* __restrict__ BT,
    void* __restrict__ Cptr, int M, int N, int K)
{
  const int lane = threadIdx.x & 31;
  const int half = lane >> 4;
  const int l16  = lane & 15;
  const int row0 = blockIdx.y * (16 * MT);
  const int col0 = blockIdx.x * (16 * NT);

  v8f acc[MT][NT];
#pragma unroll
  for (int mi = 0; mi < MT; ++mi)
#pragma unroll
    for (int t = 0; t < NT; ++t)
      acc[mi][t] = (v8f){0.f,0.f,0.f,0.f,0.f,0.f,0.f,0.f};

  for (int kb = 0; kb < K; kb += 32) {
    // A operands, 16x32 bf16 A in this code's k order:
    //   elems 0..15 : K = kb + 16*half + e (contiguous, matches B)
    v16bf av[MT];
#pragma unroll
    for (int mi = 0; mi < MT; ++mi) {
      const unsigned short* p = A + (size_t)(row0 + 16 * mi + l16) * K + kb;
      if (kb + 32 < K) __builtin_prefetch(p + 32, 0, 0);   // global_prefetch_b8
      const v8u lo = *(const v8u*)(p + 16 * half);
      const v8u hi = *(const v8u*)(p + 16 * half + 8);
      v16u au;
#pragma unroll
      for (int e = 0; e < 8; ++e) { au[e] = lo[e]; au[8 + e] = hi[e]; }
      av[mi] = as_bf16(au);
    }
    // B operands, 32x16 bf16 B in the same k order as A: lane holds column n = l16,
    // K = kb + 16*half + e  -> contiguous 32B from transposed weights
#pragma unroll
    for (int t = 0; t < NT; ++t) {
      const v16u bu =
          *(const v16u*)(BT + (size_t)(col0 + 16 * t + l16) * K + kb + 16 * half);
      const v16bf bv = as_bf16(bu);
#pragma unroll
      for (int mi = 0; mi < MT; ++mi)
        acc[mi][t] = __builtin_amdgcn_wmma_f32_16x16x32_bf16(
            false, av[mi], false, bv, (short)0, acc[mi][t], false, false);
    }
  }

  // store: C layout — lane holds (m = r + 8*half, n = l16) for r = 0..7
#pragma unroll
  for (int mi = 0; mi < MT; ++mi)
#pragma unroll
    for (int t = 0; t < NT; ++t) {
      const int col = col0 + 16 * t + l16;
#pragma unroll
      for (int r = 0; r < 8; ++r) {
        const int   m = row0 + 16 * mi + r + 8 * half;
        const float v = acc[mi][t][r];
        if constexpr (OUT_MODE == 0) {
          ((unsigned short*)Cptr)[(size_t)m * N + col] = f32_to_bf16(v);
        } else if constexpr (OUT_MODE == 1) {
          ((unsigned short*)Cptr)[(size_t)col * M + m] = f32_to_bf16(v);
        } else {
          ((float*)Cptr)[(size_t)m * N + col] = v;
        }
      }
    }
}

// ---------------------------------------------------------------------------
// Flash attention (causal, GQA rep=4), one wave per (b, h, 16-query block).
//   Q  : [B*T, 896]  bf16   K : [B*T, 224] bf16
//   Vt : [224, B*T]  bf16   O : [B*T, 896] bf16
// ---------------------------------------------------------------------------
__global__ __launch_bounds__(32) void flash_attn_wmma(
    const unsigned short* __restrict__ Qb,
    const unsigned short* __restrict__ Kb,
    const unsigned short* __restrict__ Vt,
    unsigned short* __restrict__ Ob,
    int T, int BT)
{
  constexpr int CQ = 896, CKV = 224, HD = 32, REP = 4;
  const int lane = threadIdx.x & 31;
  const int half = lane >> 4;
  const int l16  = lane & 15;
  const int qb   = blockIdx.x * 16;
  const int h    = blockIdx.y;
  const int b    = blockIdx.z;
  const int hkv  = h / REP;
  const float scale = 0.17677669529663687f;  // 1/sqrt(32)

  v16u qu;
  {
    const unsigned short* qrow = Qb + (size_t)(b * T + qb + l16) * CQ + h * HD;
    const v8u lo = *(const v8u*)(qrow + 16 * half);
    const v8u hi = *(const v8u*)(qrow + 16 * half + 8);
#pragma unroll
    for (int e = 0; e < 8; ++e) { qu[e] = lo[e]; qu[8 + e] = hi[e]; }
  }
  const v16bf qa = as_bf16(qu);

  v8f accA = (v8f){0.f,0.f,0.f,0.f,0.f,0.f,0.f,0.f};
  v8f accB = (v8f){0.f,0.f,0.f,0.f,0.f,0.f,0.f,0.f};
  float rm[8], rl[8];
#pragma unroll
  for (int r = 0; r < 8; ++r) { rm[r] = -__builtin_inff(); rl[r] = 0.f; }

  __shared__ unsigned short ptile[16 * 32];  // single-wave workgroup

  for (int jb = 0; jb < qb + 16; jb += 32) {  // jb <= qb always
    const unsigned short* kp0 =
        Kb + (size_t)(b * T + jb + l16) * CKV + hkv * HD + 16 * half;
    const v16u k0u = *(const v16u*)kp0;
    const v16u k1u = *(const v16u*)(kp0 + (size_t)16 * CKV);

    const v8f z = (v8f){0.f,0.f,0.f,0.f,0.f,0.f,0.f,0.f};
    v8f s0 = __builtin_amdgcn_wmma_f32_16x16x32_bf16(false, qa, false, as_bf16(k0u),
                                                     (short)0, z, false, false);
    v8f s1 = __builtin_amdgcn_wmma_f32_16x16x32_bf16(false, qa, false, as_bf16(k1u),
                                                     (short)0, z, false, false);

#pragma unroll
    for (int r = 0; r < 8; ++r) {
      const int q = qb + r + 8 * half;
      float v0 = s0[r] * scale;
      float v1 = s1[r] * scale;
      if (jb + l16 > q)      v0 = -__builtin_inff();
      if (jb + 16 + l16 > q) v1 = -__builtin_inff();

      float mx = fmaxf(v0, v1);
      mx = fmaxf(mx, __shfl_xor(mx, 1));
      mx = fmaxf(mx, __shfl_xor(mx, 2));
      mx = fmaxf(mx, __shfl_xor(mx, 4));
      mx = fmaxf(mx, __shfl_xor(mx, 8));
      const float mnew  = fmaxf(rm[r], mx);
      const float alpha = __expf(rm[r] - mnew);
      rm[r] = mnew;

      const float p0 = __expf(v0 - mnew);
      const float p1 = __expf(v1 - mnew);
      float rs = p0 + p1;
      rs += __shfl_xor(rs, 1);
      rs += __shfl_xor(rs, 2);
      rs += __shfl_xor(rs, 4);
      rs += __shfl_xor(rs, 8);
      rl[r] = rl[r] * alpha + rs;
      accA[r] *= alpha;
      accB[r] *= alpha;

      ptile[(r + 8 * half) * 32 + l16]      = f32_to_bf16(p0);
      ptile[(r + 8 * half) * 32 + 16 + l16] = f32_to_bf16(p1);
    }
    __syncthreads();

    v16u pu;
    {
      const unsigned short* pr = &ptile[l16 * 32];
      const v8u lo = *(const v8u*)(pr + 16 * half);
      const v8u hi = *(const v8u*)(pr + 16 * half + 8);
#pragma unroll
      for (int e = 0; e < 8; ++e) { pu[e] = lo[e]; pu[8 + e] = hi[e]; }
    }
    __syncthreads();
    const v16bf pa = as_bf16(pu);

    const unsigned short* vp0 =
        Vt + (size_t)(hkv * HD + l16) * BT + b * T + jb + 16 * half;
    const unsigned short* vp1 =
        Vt + (size_t)(hkv * HD + 16 + l16) * BT + b * T + jb + 16 * half;
    accA = __builtin_amdgcn_wmma_f32_16x16x32_bf16(false, pa, false,
                                                   as_bf16(*(const v16u*)vp0),
                                                   (short)0, accA, false, false);
    accB = __builtin_amdgcn_wmma_f32_16x16x32_bf16(false, pa, false,
                                                   as_bf16(*(const v16u*)vp1),
                                                   (short)0, accB, false, false);
  }

  unsigned short* orow = Ob + (size_t)(b * T + qb) * CQ + h * HD;
#pragma unroll
  for (int r = 0; r < 8; ++r) {
    const int   m   = r + 8 * half;
    const float inv = 1.f / rl[r];
    orow[(size_t)m * CQ + l16]      = f32_to_bf16(accA[r] * inv);
    orow[(size_t)m * CQ + 16 + l16] = f32_to_bf16(accB[r] * inv);
  }
}

// ---------------------------------------------------------------------------
extern "C" void kernel_launch(void* const* d_in, const int* in_sizes, int n_in,
                              void* d_out, int out_size, void* d_ws, size_t ws_size,
                              hipStream_t stream)
{
  const float* x  = (const float*)d_in[0];
  const float* Wq = (const float*)d_in[1];
  const float* Wk = (const float*)d_in[2];
  const float* Wv = (const float*)d_in[3];
  const float* Wo = (const float*)d_in[4];

  const int B = 4, T = 2048, C = 896, CKV = 224, H = 28;
  const int M = B * T;  // 8192

  unsigned short* ws = (unsigned short*)d_ws;
  size_t off = 0;
  unsigned short* xb  = ws + off; off += (size_t)M * C;     // bf16 x; reused as Ob
  unsigned short* Qb  = ws + off; off += (size_t)M * C;
  unsigned short* Kb  = ws + off; off += (size_t)M * CKV;
  unsigned short* VT  = ws + off; off += (size_t)M * CKV;
  unsigned short* WqT = ws + off; off += (size_t)C * C;
  unsigned short* WkT = ws + off; off += (size_t)CKV * C;
  unsigned short* WvT = ws + off; off += (size_t)CKV * C;
  unsigned short* WoT = ws + off; off += (size_t)C * C;
  unsigned short* Ob  = xb;  // x is dead after the projections (stream-ordered)

  // ---- pack inputs/weights to bf16 (weights transposed to [N,K])
  cvt_f32_to_bf16_vec<<<(M * C / 4 + 255) / 256, 256, 0, stream>>>(x, xb, M * C / 4);
  transpose_cvt_bf16<<<(C * C   + 255) / 256, 256, 0, stream>>>(Wq, WqT, C, C);
  transpose_cvt_bf16<<<(C * CKV + 255) / 256, 256, 0, stream>>>(Wk, WkT, C, CKV);
  transpose_cvt_bf16<<<(C * CKV + 255) / 256, 256, 0, stream>>>(Wv, WvT, C, CKV);
  transpose_cvt_bf16<<<(C * C   + 255) / 256, 256, 0, stream>>>(Wo, WoT, C, C);

  const dim3 blk(32);

  // ---- projections (32x112 tile per wave; V written transposed)
  gemm_bf16_wmma<0, 2, 7><<<dim3(C / 112,   M / 32), blk, 0, stream>>>(xb, WqT, Qb, M, C,   C);
  gemm_bf16_wmma<0, 2, 7><<<dim3(CKV / 112, M / 32), blk, 0, stream>>>(xb, WkT, Kb, M, CKV, C);
  gemm_bf16_wmma<1, 2, 7><<<dim3(CKV / 112, M / 32), blk, 0, stream>>>(xb, WvT, VT, M, CKV, C);

  // ---- causal GQA flash attention
  flash_attn_wmma<<<dim3(T / 16, H, B), blk, 0, stream>>>(Qb, Kb, VT, Ob, T, M);

  // ---- output projection (bf16 in, fp32 out)
  gemm_bf16_wmma<2, 2, 7><<<dim3(C / 112, M / 32), blk, 0, stream>>>(Ob, WoT, (float*)d_out, M, C, C);
}
